// CustomCLIP_29197187678946
// MI455X (gfx1250) — matrix operand order repaired and checked by reading, not compiled, not run on hardware
//
#include <hip/hip_runtime.h>

typedef __attribute__((ext_vector_type(16))) _Float16 v16h;
typedef __attribute__((ext_vector_type(8)))  _Float16 v8h;
typedef __attribute__((ext_vector_type(8)))  float    v8f;

#define B_   32
#define N_   196
#define D_   512
#define C_   500
#define ND_  51
#define TOPK 50
#define MT_  208   // m padded to 13*16 for WMMA tiles
#define MPAD 224   // m padded to 7*32 for the top-k scan
#define CT   16    // columns (c) per workgroup in the big kernel

// ---------------- helpers ----------------
__device__ __forceinline__ float dot512(const float* __restrict__ a,
                                        const float* __restrict__ b, int lane) {
  float s = 0.f;
  #pragma unroll
  for (int base = 0; base < D_; base += 128) {
    const float4 x = *(const float4*)(a + base + lane * 4);
    const float4 y = *(const float4*)(b + base + lane * 4);
    s += x.x * y.x + x.y * y.y + x.z * y.z + x.w * y.w;
  }
  #pragma unroll
  for (int off = 16; off > 0; off >>= 1) s += __shfl_xor(s, off, 32);
  return s;
}

// ---------------- f32 -> f16 staging (one-time per call) ----------------
__global__ __launch_bounds__(256) void k_convA(const float* __restrict__ src,
                                               _Float16* __restrict__ dst) {
  // src [32][196][512] -> dst [32][208][512], pad rows zero
  int i = blockIdx.x * 256 + threadIdx.x;
  if (i >= B_ * MT_ * (D_ / 8)) return;
  int b  = i / (MT_ * (D_ / 8));
  int r  = (i / (D_ / 8)) % MT_;
  int k8 = (i & (D_ / 8 - 1)) * 8;
  v8h h;
  if (r < N_) {
    const float* s = src + ((size_t)b * N_ + r) * D_ + k8;
    #pragma unroll
    for (int e = 0; e < 8; ++e) h[e] = (_Float16)s[e];
  } else {
    #pragma unroll
    for (int e = 0; e < 8; ++e) h[e] = (_Float16)0.f;
  }
  *(v8h*)(dst + ((size_t)b * MT_ + r) * D_ + k8) = h;
}

__global__ __launch_bounds__(256) void k_convT(const float* __restrict__ src,
                                               _Float16* __restrict__ dst) {
  int i = blockIdx.x * 256 + threadIdx.x;
  if (i >= ND_ * C_ * (D_ / 8)) return;
  size_t off = (size_t)i * 8;
  const float* s = src + off;
  v8h h;
  #pragma unroll
  for (int e = 0; e < 8; ++e) h[e] = (_Float16)s[e];
  *(v8h*)(dst + off) = h;
}

// ---------------- v logits + softmax over n ----------------
__global__ __launch_bounds__(256) void k_vlog(const float* __restrict__ mean_t,
                                              const float* __restrict__ all_t,
                                              float* __restrict__ vlog) {
  int gid = blockIdx.x * 256 + threadIdx.x;
  int wid = gid >> 5, lane = gid & 31;
  if (wid >= ND_ * C_) return;
  int n = wid / C_, c = wid % C_;
  float d = dot512(mean_t + (size_t)c * D_, all_t + ((size_t)n * C_ + c) * D_, lane);
  if (lane == 0) vlog[wid] = d;  // [n][c]
}

__global__ __launch_bounds__(256) void k_softmax_v(const float* __restrict__ vlog,
                                                   float* __restrict__ vt) {
  int c = blockIdx.x * 256 + threadIdx.x;
  if (c >= C_) return;
  float mx = -1e30f;
  for (int n = 0; n < ND_; ++n) mx = fmaxf(mx, vlog[n * C_ + c]);
  float s = 0.f;
  for (int n = 0; n < ND_; ++n) s += expf(vlog[n * C_ + c] - mx);
  float inv = 1.f / s;
  for (int n = 0; n < ND_; ++n) vt[c * ND_ + n] = expf(vlog[n * C_ + c] - mx) * inv;
}

// ---------------- w[b,m] ----------------
__global__ __launch_bounds__(256) void k_w(const float* __restrict__ img,
                                           const float* __restrict__ loc,
                                           float* __restrict__ wW) {
  int gid = blockIdx.x * 256 + threadIdx.x;
  int wid = gid >> 5, lane = gid & 31;
  if (wid >= B_ * N_) return;
  int b = wid / N_, m = wid % N_;
  float d = dot512(img + (size_t)b * D_, loc + ((size_t)b * N_ + m) * D_, lane);
  if (lane == 0) wW[wid] = d;
}

// ---------------- wg: top-50 indices of sim[b,0,0,:] (exact f32), softmax of gathered w ----------
__global__ __launch_bounds__(256) void k_wg(const float* __restrict__ loc,
                                            const float* __restrict__ all_t,
                                            const float* __restrict__ wW,
                                            float* __restrict__ wgW) {
  __shared__ float sLDS[MPAD];
  __shared__ int   selLDS[TOPK];
  __shared__ float tmpLDS[TOPK];
  int b = blockIdx.x;
  int tid = threadIdx.x, lane = tid & 31, wave = tid >> 5;
  const float* t00 = all_t;  // all_text[0][0][:]
  for (int m = wave; m < N_; m += 8) {
    float d = dot512(loc + ((size_t)b * N_ + m) * D_, t00, lane);
    if (lane == 0) sLDS[m] = d;
  }
  if (tid >= N_ && tid < MPAD) sLDS[tid] = -1e30f;
  __syncthreads();
  if (wave == 0) {
    float vv[7];
    #pragma unroll
    for (int j = 0; j < 7; ++j) vv[j] = sLDS[lane + 32 * j];
    for (int k = 0; k < TOPK; ++k) {
      float bv = vv[0]; int bj = 0;
      #pragma unroll
      for (int j = 1; j < 7; ++j) if (vv[j] > bv) { bv = vv[j]; bj = j; }
      int bm = lane + 32 * bj;
      #pragma unroll
      for (int off = 16; off > 0; off >>= 1) {
        float ov = __shfl_xor(bv, off, 32);
        int   om = __shfl_xor(bm, off, 32);
        if (ov > bv || (ov == bv && om < bm)) { bv = ov; bm = om; }
      }
      int rl = bm & 31, rj = bm >> 5;
      #pragma unroll
      for (int j = 0; j < 7; ++j) if (j == rj && lane == rl) vv[j] = -1e30f;
      if (lane == 0) selLDS[k] = bm;
    }
  }
  __syncthreads();
  if (tid == 0) {
    float mx = -1e30f;
    for (int k = 0; k < TOPK; ++k) { float x = wW[b * N_ + selLDS[k]]; tmpLDS[k] = x; mx = fmaxf(mx, x); }
    float s = 0.f;
    for (int k = 0; k < TOPK; ++k) { float e = expf(tmpLDS[k] - mx); tmpLDS[k] = e; s += e; }
    float inv = 1.f / s;
    for (int k = 0; k < TOPK; ++k) wgW[b * TOPK + k] = tmpLDS[k] * inv;
  }
}

// ---------------- base logits ----------------
__global__ __launch_bounds__(256) void k_base(const float* __restrict__ img,
                                              const float* __restrict__ mean_t,
                                              float* __restrict__ out) {
  int gid = blockIdx.x * 256 + threadIdx.x;
  int wid = gid >> 5, lane = gid & 31;
  if (wid >= B_ * C_) return;
  int b = wid / C_, c = wid % C_;
  float d = dot512(img + (size_t)b * D_, mean_t + (size_t)c * D_, lane);
  if (lane == 0) out[wid] = d;
}

// ---------------- the big one: WMMA sim tiles + sorted top-k + bias accumulation ----------
__global__ __launch_bounds__(256) void k_bias(const _Float16* __restrict__ A16,  // [32][208][512]
                                              const _Float16* __restrict__ T16,  // [51][500][512]
                                              const float* __restrict__ vt,      // [500][51]
                                              const float* __restrict__ wgW,     // [32][50]
                                              float* __restrict__ out) {         // [32][500]
  __shared__ float simLDS[CT * MPAD];  // [col][m]
  __shared__ float wgLDS[TOPK];
  const int b   = blockIdx.y;
  const int c0  = blockIdx.x * CT;
  const int tid = threadIdx.x;
  const int lane = tid & 31, wave = tid >> 5;
  const int lh = lane >> 4, l15 = lane & 15;

  if (tid < TOPK) wgLDS[tid] = wgW[b * TOPK + tid];

  float bias0 = 0.f, bias1 = 0.f;
  const _Float16* arowBase = A16 + ((size_t)b * MT_ + l15) * D_;
  const int cb = (c0 + l15 < C_) ? (c0 + l15) : (C_ - 1);  // clamp pad columns

  for (int n = 0; n < ND_; ++n) {
    const _Float16* bcol = T16 + ((size_t)n * C_ + cb) * D_;
    // --- GEMM: 13 m-tiles over 8 waves, K=512 in 16 WMMA steps ---
    for (int j = wave; j < MT_ / 16; j += 8) {
      const int m0 = j * 16;
      const _Float16* arow = arowBase + (size_t)m0 * D_;
      v8f acc = {};
      #pragma unroll
      for (int kc = 0; kc < 16; ++kc) {
        const int kb = kc * 32;
        // A 16x32 f16 layout: lane-half 0 -> K {0..7,16..23}, half 1 -> +8
        v8h a0 = *(const v8h*)(arow + kb + 8 * lh);
        v8h a1 = *(const v8h*)(arow + kb + 16 + 8 * lh);
        // B 32x16 f16 layout: lane-half 0 -> K {0..7,16..23}, half 1 -> +8
        v8h b0 = *(const v8h*)(bcol + kb + 8 * lh);
        v8h b1 = *(const v8h*)(bcol + kb + 16 + 8 * lh);
        v16h av, bv;
        #pragma unroll
        for (int e = 0; e < 8; ++e) { av[e] = a0[e]; av[e + 8] = a1[e];
                                      bv[e] = b0[e]; bv[e + 8] = b1[e]; }
        acc = __builtin_amdgcn_wmma_f32_16x16x32_f16(false, av, false, bv,
                                                     (short)0, acc, false, false);
      }
      // D layout: element r -> M = r + 8*laneHalf, N = lane&15
      float* dst = &simLDS[l15 * MPAD + m0 + 8 * lh];
      #pragma unroll
      for (int r = 0; r < 8; ++r) dst[r] = acc[r];
    }
    if (n + 1 < ND_) {  // pull next n's B columns toward the caches
      const _Float16* p = T16 + ((size_t)(n + 1) * C_ + min(c0 + (tid >> 4), C_ - 1)) * D_
                          + (tid & 15) * 32;
      __builtin_prefetch((const void*)p, 0, 3);
    }
    __syncthreads();
    // mask pad rows m in [196, 224)
    for (int i = tid; i < (MPAD - N_) * CT; i += 256)
      simLDS[(i & (CT - 1)) * MPAD + N_ + (i >> 4)] = -1e30f;
    __syncthreads();
    // --- sorted top-50 per column, 2 columns per wave ---
    #pragma unroll
    for (int cc = 0; cc < 2; ++cc) {
      const int col = wave * 2 + cc;
      const float* sc = &simLDS[col * MPAD];
      float vv[7];
      #pragma unroll
      for (int j = 0; j < 7; ++j) vv[j] = sc[lane + 32 * j];
      float colsum = 0.f;
      for (int k = 0; k < TOPK; ++k) {
        float bv = vv[0]; int bj = 0;
        #pragma unroll
        for (int j = 1; j < 7; ++j) if (vv[j] > bv) { bv = vv[j]; bj = j; }
        int bm = lane + 32 * bj;
        #pragma unroll
        for (int off = 16; off > 0; off >>= 1) {   // argmax, min-index tie-break
          float ov = __shfl_xor(bv, off, 32);
          int   om = __shfl_xor(bm, off, 32);
          if (ov > bv || (ov == bv && om < bm)) { bv = ov; bm = om; }
        }
        int rl = bm & 31, rj = bm >> 5;
        #pragma unroll
        for (int j = 0; j < 7; ++j) if (j == rj && lane == rl) vv[j] = -1e30f;
        colsum += wgLDS[k] * bv;
      }
      const int c = c0 + col;
      const float vnc = (c < C_) ? vt[c * ND_ + n] : 0.f;
      if (cc == 0) bias0 += vnc * colsum; else bias1 += vnc * colsum;
    }
    __syncthreads();
  }
  if (lane == 0) {
    int cA = c0 + wave * 2;     if (cA < C_) out[b * C_ + cA] += bias0;
    int cB = c0 + wave * 2 + 1; if (cB < C_) out[b * C_ + cB] += bias1;
  }
}

// ---------------- launch ----------------
extern "C" void kernel_launch(void* const* d_in, const int* in_sizes, int n_in,
                              void* d_out, int out_size, void* d_ws, size_t ws_size,
                              hipStream_t stream) {
  (void)in_sizes; (void)n_in; (void)out_size; (void)ws_size;
  const float* img  = (const float*)d_in[0];   // [32][512]
  const float* loc  = (const float*)d_in[1];   // [32][196][512]
  const float* txt  = (const float*)d_in[2];   // [51][500][512]
  const float* mean = (const float*)d_in[3];   // [500][512]
  float* out = (float*)d_out;                  // [32][500]
  char* ws = (char*)d_ws;

  const size_t offA16  = 0;
  const size_t offT16  = offA16 + (size_t)B_ * MT_ * D_ * 2;      // 6.8 MB
  const size_t offVlog = offT16 + (size_t)ND_ * C_ * D_ * 2;      // +26.1 MB
  const size_t offVt   = offVlog + (size_t)ND_ * C_ * 4;
  const size_t offW    = offVt + (size_t)C_ * ND_ * 4;
  const size_t offWg   = offW + (size_t)B_ * N_ * 4;

  _Float16* A16 = (_Float16*)(ws + offA16);
  _Float16* T16 = (_Float16*)(ws + offT16);
  float* vlog = (float*)(ws + offVlog);
  float* vt   = (float*)(ws + offVt);
  float* wW   = (float*)(ws + offW);
  float* wgW  = (float*)(ws + offWg);

  k_convA<<<(B_ * MT_ * (D_ / 8) + 255) / 256, 256, 0, stream>>>(loc, A16);
  k_convT<<<(ND_ * C_ * (D_ / 8) + 255) / 256, 256, 0, stream>>>(txt, T16);
  k_vlog<<<(ND_ * C_ + 7) / 8, 256, 0, stream>>>(mean, txt, vlog);
  k_softmax_v<<<(C_ + 255) / 256, 256, 0, stream>>>(vlog, vt);
  k_w<<<(B_ * N_ + 7) / 8, 256, 0, stream>>>(img, loc, wW);
  k_wg<<<B_, 256, 0, stream>>>(loc, txt, wW, wgW);
  k_base<<<(B_ * C_ + 7) / 8, 256, 0, stream>>>(img, mean, out);
  dim3 grid((C_ + CT - 1) / CT, B_);  // 32 x 32 workgroups
  k_bias<<<grid, 256, 0, stream>>>(A16, T16, vt, wgW, out);
}
